// ISTFT_48258252538127
// MI455X (gfx1250) — matrix operand order repaired and checked by reading, not compiled, not run on hardware
//
#include <hip/hip_runtime.h>

typedef __attribute__((ext_vector_type(16))) _Float16 v16h;
typedef __attribute__((ext_vector_type(8)))  _Float16 v8h;
typedef __attribute__((ext_vector_type(8)))  float    v8f;

#define NFFT   2048
#define NHALF  1024
#define HOP    512
#define NB     8
#define NT     1024     // frames
#define NFREQ  1025     // NFFT/2 + 1
#define KTOT   4096     // real(2048) + imag(2048) stacked on K
#define BM     128
#define BN     128
#define BK     64
#define LDA    (BK + 8) // halves; row stride 144 B (16B multiple)
#define LDB    (BK + 8)
#define NROUND (KTOT / BK)

// ws layout (bytes)
#define SBUF_OFF   0u                          // f32 s[b][t][o] : 64 MB
#define F16_OFF    (size_t)(64u * 1024 * 1024) // f16 full16[b][t][k] : 64 MB
#define AHT_OFF    (size_t)(128u * 1024 * 1024)// f16 AhT[o][k] : 16 MB

// --------------------------------------------------------------------------
// Prepack 1: AhT[o][k] = (k<2048 ? Wr[k][o] : -Wi[k-2048][o]) as f16
// --------------------------------------------------------------------------
__global__ __launch_bounds__(256) void pack_weights_kernel(
    const float* __restrict__ Wr, const float* __restrict__ Wi,
    _Float16* __restrict__ AhT)
{
    __shared__ _Float16 tile[32][33];
    const int k0 = blockIdx.x * 32;
    const int o0 = blockIdx.y * 32;
    const int tx = threadIdx.x & 31;
    const int ty = threadIdx.x >> 5;          // 0..7
    #pragma unroll
    for (int i = 0; i < 4; ++i) {
        const int k = k0 + ty + 8 * i;
        const float v = (k < NFFT) ? Wr[(size_t)k * NFFT + o0 + tx]
                                   : -Wi[(size_t)(k - NFFT) * NFFT + o0 + tx];
        tile[ty + 8 * i][tx] = (_Float16)v;
    }
    __syncthreads();
    #pragma unroll
    for (int i = 0; i < 4; ++i)
        AhT[(size_t)(o0 + ty + 8 * i) * KTOT + k0 + tx] = tile[tx][ty + 8 * i];
}

// --------------------------------------------------------------------------
// Prepack 2: full16[b][t][k] = conjugate-symmetric spectrum (real||imag), f16
// --------------------------------------------------------------------------
__global__ __launch_bounds__(256) void pack_spectrum_kernel(
    const float* __restrict__ realp, const float* __restrict__ imagp,
    _Float16* __restrict__ full16)
{
    const size_t gid = (size_t)blockIdx.x * 256 + threadIdx.x; // < NB*NT*KTOT
    const int    k   = (int)(gid & (KTOT - 1));
    const size_t bt  = gid >> 12;                              // b*NT + t
    float v;
    if (k < NFFT) {
        const int f = (k <= NHALF) ? k : (NFFT - k);
        v = realp[bt * NFREQ + f];
    } else {
        const int i = k - NFFT;
        v = (i <= NHALF) ? imagp[bt * NFREQ + i] : -imagp[bt * NFREQ + (NFFT - i)];
    }
    full16[gid] = (_Float16)v;
}

// --------------------------------------------------------------------------
// GEMM: s[b][t][o] = sum_k AhT[o][k] * full16[b][t][k]
// block 256 (8 waves), tile 128(M=o) x 128(N=t), BK=64, wave tile 32x64,
// register-pipelined global->LDS staging, direct float4 epilogue stores
// --------------------------------------------------------------------------
__global__ __launch_bounds__(256) void istft_gemm_kernel(
    const _Float16* __restrict__ AhT, const _Float16* __restrict__ full16,
    float* __restrict__ sbuf)
{
    __shared__ __align__(16) _Float16 As[BM][LDA];   // 18,432 B
    __shared__ __align__(16) _Float16 Bs[BN][LDB];   // 18,432 B

    const int tid = threadIdx.x;
    const int t0  = blockIdx.x * BN;
    const int o0  = blockIdx.y * BM;
    const int bb  = blockIdx.z;

    const int wave = tid >> 5;
    const int lane = tid & 31;
    const int r    = lane & 15;
    const int g    = lane >> 4;
    const int wm   = (wave & 3) * 32;     // 4 waves along M
    const int wn   = (wave >> 2) * 64;    // 2 waves along N

    v8f acc[2][4] = {};                   // 32(M) x 64(N) per wave

    // loaders: each tile is 128 rows x 64 k-halves; 32 halves (4 x v8h)/thread
    const int a_m  = tid >> 1;            // 0..127
    const int a_kq = (tid & 1) * 32;      // 0 or 32
    const int b_n  = tid >> 1;            // 0..127
    const int b_kq = (tid & 1) * 32;      // 0 or 32

    const size_t aBase = (size_t)(o0 + a_m) * KTOT + a_kq;
    const size_t bBase = ((size_t)bb * NT + (t0 + b_n)) * KTOT + b_kq;

    v8h areg[4], breg[4];
    #pragma unroll
    for (int j = 0; j < 4; ++j) areg[j] = *(const v8h*)(AhT + aBase + 8 * j);
    #pragma unroll
    for (int j = 0; j < 4; ++j) breg[j] = *(const v8h*)(full16 + bBase + 8 * j);

    for (int rnd = 0; rnd < NROUND; ++rnd) {
        __syncthreads();                  // previous round's fragment reads done
        #pragma unroll
        for (int j = 0; j < 4; ++j) *(v8h*)&As[a_m][a_kq + 8 * j] = areg[j];
        #pragma unroll
        for (int j = 0; j < 4; ++j) *(v8h*)&Bs[b_n][b_kq + 8 * j] = breg[j];
        __syncthreads();

        if (rnd + 1 < NROUND) {           // prefetch next round into registers
            const size_t koff = (size_t)(rnd + 1) * BK;
            #pragma unroll
            for (int j = 0; j < 4; ++j)
                areg[j] = *(const v8h*)(AhT + aBase + koff + 8 * j);
            #pragma unroll
            for (int j = 0; j < 4; ++j)
                breg[j] = *(const v8h*)(full16 + bBase + koff + 8 * j);
        }

        #pragma unroll
        for (int ks = 0; ks < 2; ++ks) {  // two 16x16x32 K-substeps per round
            const int koff = ks * 32;
            v16h afr[2], bfr[4];
            #pragma unroll
            for (int mi = 0; mi < 2; ++mi) {
                const _Float16* Ar = &As[wm + mi * 16 + r][koff];
                v8h lo = *(const v8h*)(Ar + g * 8);       // K = koff + g*8 ..
                v8h hi = *(const v8h*)(Ar + 16 + g * 8);  // K = koff+16+g*8 ..
                #pragma unroll
                for (int e = 0; e < 8; ++e) { afr[mi][e] = lo[e]; afr[mi][e + 8] = hi[e]; }
            }
            #pragma unroll
            for (int ni = 0; ni < 4; ++ni) {
                const _Float16* Br = &Bs[wn + ni * 16 + r][koff];
                v8h lo = *(const v8h*)(Br + g * 8);
                v8h hi = *(const v8h*)(Br + 16 + g * 8);
                #pragma unroll
                for (int e = 0; e < 8; ++e) { bfr[ni][e] = lo[e]; bfr[ni][e + 8] = hi[e]; }
            }
            #pragma unroll
            for (int mi = 0; mi < 2; ++mi)
                #pragma unroll
                for (int ni = 0; ni < 4; ++ni)
                    acc[mi][ni] = __builtin_amdgcn_wmma_f32_16x16x32_f16(
                        false, afr[mi], false, bfr[ni],
                        (short)0, acc[mi][ni], false, false);
        }
    }

    // epilogue: D elements v=0..7 are 8 consecutive o -> two float4 per tile
    #pragma unroll
    for (int mi = 0; mi < 2; ++mi) {
        #pragma unroll
        for (int ni = 0; ni < 4; ++ni) {
            const int t = t0 + wn + ni * 16 + r;
            const int o = o0 + wm + mi * 16 + 8 * g;
            float* dst = sbuf + ((size_t)bb * NT + t) * NFFT + o;
            float4 lo = { acc[mi][ni][0], acc[mi][ni][1], acc[mi][ni][2], acc[mi][ni][3] };
            float4 hi = { acc[mi][ni][4], acc[mi][ni][5], acc[mi][ni][6], acc[mi][ni][7] };
            *(float4*)(dst)     = lo;
            *(float4*)(dst + 4) = hi;
        }
    }
}

// --------------------------------------------------------------------------
// Overlap-add + window-sum normalization + trim (center=True)
// --------------------------------------------------------------------------
__global__ __launch_bounds__(256) void istft_ola_kernel(
    const float* __restrict__ sbuf, const float* __restrict__ ola,
    float* __restrict__ out, int length)
{
    const long gid   = (long)blockIdx.x * blockDim.x + threadIdx.x;
    const long total = (long)NB * length;
    if (gid >= total) return;
    const int b = (int)(gid / length);
    const int j = (int)(gid - (long)b * length);
    const int n = j + NFFT / 2;

    int tmax = n / HOP;                              if (tmax > NT - 1) tmax = NT - 1;
    int tmin = (n - (NFFT - 1) + (HOP - 1)) / HOP;   if (tmin < 0) tmin = 0;

    float y = 0.0f, ws = 0.0f;
    for (int t = tmin; t <= tmax; ++t) {
        const int o = n - t * HOP;
        y  += sbuf[((size_t)b * NT + t) * NFFT + o];
        ws += ola[o];
    }
    out[(size_t)b * length + j] = y / fmaxf(ws, 1e-11f);
}

// --------------------------------------------------------------------------
extern "C" void kernel_launch(void* const* d_in, const int* in_sizes, int n_in,
                              void* d_out, int out_size, void* d_ws, size_t ws_size,
                              hipStream_t stream) {
    const float* realp = (const float*)d_in[0];
    const float* imagp = (const float*)d_in[1];
    const float* Wr    = (const float*)d_in[2];
    const float* Wi    = (const float*)d_in[3];
    const float* ola   = (const float*)d_in[4];
    const int length   = (NT - 1) * HOP;             // 523776

    char* ws = (char*)d_ws;
    float*     sbuf   = (float*)(ws + SBUF_OFF);
    _Float16*  full16 = (_Float16*)(ws + F16_OFF);
    _Float16*  AhT    = (_Float16*)(ws + AHT_OFF);

    // prepack (one-shot, bandwidth-trivial)
    dim3 gw(KTOT / 32, NFFT / 32);                   // 128 x 64
    pack_weights_kernel<<<gw, 256, 0, stream>>>(Wr, Wi, AhT);

    const size_t specElems = (size_t)NB * NT * KTOT; // 33,554,432
    pack_spectrum_kernel<<<(unsigned)(specElems / 256), 256, 0, stream>>>(
        realp, imagp, full16);

    // main WMMA GEMM
    dim3 gg(NT / BN, NFFT / BM, NB);                 // 8 x 16 x 8
    istft_gemm_kernel<<<gg, 256, 0, stream>>>(AhT, full16, sbuf);

    // overlap-add + normalize + trim
    const long total = (long)NB * length;
    const int blocks = (int)((total + 255) / 256);
    istft_ola_kernel<<<blocks, 256, 0, stream>>>(sbuf, ola, (float*)d_out, length);
}
